// MultiHeadSpectralAttention_36962488550069
// MI455X (gfx1250) — matrix operand order repaired and checked by reading, not compiled, not run on hardware
//
#include <hip/hip_runtime.h>
#include <hip/hip_bf16.h>
#include <math.h>

typedef __attribute__((ext_vector_type(16))) _Float16 v16h;
typedef __attribute__((ext_vector_type(8)))  float    v8f;

#define B_   8
#define N_   4096
#define D_   1024
#define H_   16
#define HD_  64
#define FB_  33      // hd/2+1
#define COLS_ 66     // re(0..32) ++ im(0..32)
#define KP_   96     // synthesis K padded to 3x32

__device__ __forceinline__ float gelu_exact(float v) {
    return 0.5f * v * (1.0f + erff(v * 0.70710678118654752f));
}

// ---------------- zero workspace region ----------------
__global__ void zero_f32(float* p, int n) {
    int i = blockIdx.x * blockDim.x + threadIdx.x;
    if (i < n) p[i] = 0.0f;
}

// ---------------- pass 1: LN stats + ctx partial sums ----------------
// one wave per token; 8 waves/block, 8 iters -> 64 tokens/block
__global__ __launch_bounds__(256) void ln_stats_kernel(
    const float* __restrict__ x, float* __restrict__ mu, float* __restrict__ rstd,
    float* __restrict__ ctxsum)
{
    __shared__ float cpart[D_];
    const int lane = threadIdx.x & 31;
    const int wave = threadIdx.x >> 5;
    const int b = blockIdx.x >> 6;             // 64 blocks per batch
    const int nbase = (blockIdx.x & 63) * 64;

    for (int i = threadIdx.x; i < D_; i += 256) cpart[i] = 0.0f;
    __syncthreads();

    float acc[32];
#pragma unroll
    for (int p = 0; p < 32; ++p) acc[p] = 0.0f;

    for (int it = 0; it < 8; ++it) {
        const int n = nbase + it * 8 + wave;
        const float* row = x + ((size_t)(b * N_ + n)) * D_;
        float vals[32];
        float s = 0.0f, s2 = 0.0f;
#pragma unroll
        for (int p = 0; p < 32; ++p) {
            float v = row[p * 32 + lane];
            vals[p] = v; s += v; s2 += v * v;
        }
#pragma unroll
        for (int off = 16; off > 0; off >>= 1) {
            s  += __shfl_xor(s,  off, 32);
            s2 += __shfl_xor(s2, off, 32);
        }
        const float m = s * (1.0f / D_);
        const float var = s2 * (1.0f / D_) - m * m;
        const float r = rsqrtf(var + 1e-5f);
        if (lane == 0) { mu[b * N_ + n] = m; rstd[b * N_ + n] = r; }
#pragma unroll
        for (int p = 0; p < 32; ++p) acc[p] += (vals[p] - m) * r;
    }
#pragma unroll
    for (int p = 0; p < 32; ++p) atomicAdd(&cpart[p * 32 + lane], acc[p]);
    __syncthreads();
    for (int i = threadIdx.x; i < D_; i += 256)
        atomicAdd(&ctxsum[b * D_ + i], cpart[i]);
}

// ---------------- pass 2a: hmid = gelu(ctx @ W1 + b1) ----------------
__global__ __launch_bounds__(256) void ctx_mlp1_kernel(
    const float* __restrict__ ctxsum, const float* __restrict__ gamma,
    const float* __restrict__ beta, const float* __restrict__ W1,
    const float* __restrict__ b1, float* __restrict__ hmid)
{
    const int i = blockIdx.x * 256 + threadIdx.x;   // 8192 total
    const int b = i >> 10;
    const int o = i & 1023;
    float acc = b1[o];
    for (int d = 0; d < D_; ++d) {
        float c = gamma[d] * ctxsum[b * D_ + d] * (1.0f / N_) + beta[d];
        acc += c * W1[d * D_ + o];
    }
    hmid[i] = gelu_exact(acc);
}

// ---------------- pass 2b: ap = hmid @ W2 + b2 -> eff filter/bias ----------------
__global__ __launch_bounds__(256) void ctx_mlp2_kernel(
    const float* __restrict__ hmid, const float* __restrict__ W2,
    const float* __restrict__ b2, const float* __restrict__ base_filter,
    const float* __restrict__ base_bias, float* __restrict__ filt,
    float* __restrict__ biasEff)
{
    const int i = blockIdx.x * 256 + threadIdx.x;
    if (i >= B_ * H_ * FB_) return;
    const int b = i / (H_ * FB_);
    const int hf = i % (H_ * FB_);                  // h*33 + k
    const int o0 = hf * 2, o1 = hf * 2 + 1;
    float s = b2[o0], t = b2[o1];
    for (int d = 0; d < D_; ++d) {
        const float hv = hmid[b * D_ + d];
        s += hv * W2[d * (H_ * FB_ * 2) + o0];
        t += hv * W2[d * (H_ * FB_ * 2) + o1];
    }
    filt[b * (H_ * FB_) + hf]    = base_filter[hf] * (1.0f + s);
    biasEff[b * (H_ * FB_) + hf] = base_bias[hf] + t;
}

// ---------------- DFT matrices in WMMA B-fragment layout ----------------
// analysis: 5 ftiles x 2 ksteps x 512 halfs ; synthesis: 4 jtiles x 3 ksteps x 512 halfs
__global__ void dft_setup_kernel(_Float16* __restrict__ gaFrag, _Float16* __restrict__ syFrag)
{
    const float PI32 = 0.0981747704246810387f;   // pi/32
    int idx = blockIdx.x * 256 + threadIdx.x;
    if (idx < 5 * 2 * 512) {
        const int frag = idx >> 9, e = idx & 511;
        const int lane = e >> 4, el = e & 15;
        const int f = frag >> 1, s = frag & 1;
        const int j = s * 32 + ((lane >= 16) ? 16 : 0) + el;  // K (sample)
        const int col = f * 16 + (lane & 15);                 // N (freq column)
        float v = 0.0f;
        if (col < FB_) {
            int t = (j * col) & 63;
            v = cosf((float)t * PI32) * 0.125f;
        } else if (col < COLS_) {
            int k = col - FB_;
            int t = (j * k) & 63;
            v = -sinf((float)t * PI32) * 0.125f;
        }
        gaFrag[idx] = (_Float16)v;
        return;
    }
    idx -= 5 * 2 * 512;
    if (idx < 4 * 3 * 512) {
        const int frag = idx >> 9, e = idx & 511;
        const int lane = e >> 4, el = e & 15;
        const int jt = frag / 3, s = frag % 3;
        const int c = s * 32 + ((lane >= 16) ? 16 : 0) + el;  // K (freq column)
        const int j = jt * 16 + (lane & 15);                  // N (sample)
        float v = 0.0f;
        if (c < COLS_) {
            const int k = (c < FB_) ? c : (c - FB_);
            const float w = (k == 0 || k == 32) ? 1.0f : 2.0f;
            const int t = (j * k) & 63;
            v = (c < FB_) ? (w * cosf((float)t * PI32) * 0.125f)
                          : (-w * sinf((float)t * PI32) * 0.125f);
        }
        syFrag[idx] = (_Float16)v;
    }
}

// ---------------- fused spectral pass: LN -> rfft(WMMA) -> mod+gelu -> irfft(WMMA) ----------------
// one wave per (b, h, 16-token tile); 4 waves/block
__global__ __launch_bounds__(128) void spectral_kernel(
    const float* __restrict__ x, const float* __restrict__ gamma,
    const float* __restrict__ beta, const float* __restrict__ mu,
    const float* __restrict__ rstd, const float* __restrict__ filt,
    const float* __restrict__ biasEff, const _Float16* __restrict__ gaFrag,
    const _Float16* __restrict__ syFrag, float* __restrict__ out)
{
    __shared__ _Float16 lds[4][16 * KP_];
    const int wave = threadIdx.x >> 5;
    const int lane = threadIdx.x & 31;
    const int item = blockIdx.x * 4 + wave;     // 32768 items
    const int tile = item & 255;
    const int h = (item >> 8) & 15;
    const int b = item >> 12;
    const int n0 = tile * 16;
    const int tok = lane & 15;
    const int hi = lane >> 4;
    const int n = n0 + tok;

    const float m_ = mu[b * N_ + n];
    const float r_ = rstd[b * N_ + n];
    const float* xrow = x + ((size_t)(b * N_ + n)) * D_ + h * HD_;
    const float* gm = gamma + h * HD_;
    const float* bt = beta + h * HD_;

    // A fragments for analysis: M=token, K=sample j
    v16h afrag[2];
#pragma unroll
    for (int s = 0; s < 2; ++s) {
#pragma unroll
        for (int g = 0; g < 2; ++g) {
            const int j0 = s * 32 + hi * 16 + g * 8;
#pragma unroll
            for (int w = 0; w < 8; ++w) {
                const int j = j0 + w;
                const float xn = (xrow[j] - m_) * r_ * gm[j] + bt[j];
                afrag[s][g * 8 + w] = (_Float16)xn;
            }
        }
    }

    _Float16* myLds = lds[wave];
    const float* fl_base = filt + (b * H_ + h) * FB_;
    const float* bi_base = biasEff + (b * H_ + h) * FB_;

    // analysis: F^T(16x80) = X^T(16x64) * G^T(64x80)
#pragma unroll
    for (int f = 0; f < 5; ++f) {
        v8f c = {};
#pragma unroll
        for (int s = 0; s < 2; ++s) {
            v16h bfr = *(const v16h*)(gaFrag + (size_t)(f * 2 + s) * 512 + lane * 16);
            c = __builtin_amdgcn_wmma_f32_16x16x32_f16(false, afrag[s], false, bfr,
                                                       (short)0, c, false, false);
        }
        const int col = f * 16 + tok;           // this lane's frequency column
        float flv = 0.0f, biv = 0.0f;
        const bool valid = col < COLS_;
        if (col < FB_)      { flv = fl_base[col];       biv = bi_base[col]; }
        else if (valid)     { flv = fl_base[col - FB_]; biv = 0.0f; }
#pragma unroll
        for (int r = 0; r < 8; ++r) {
            const float v = c[r] * flv + biv;
            const float gv = valid ? gelu_exact(v) : 0.0f;
            const int mrow = r + hi * 8;
            myLds[mrow * KP_ + col] = (_Float16)gv;
        }
    }
    // zero pad columns 80..95
#pragma unroll
    for (int q = 0; q < 8; ++q) {
        const int e = lane * 8 + q;             // 0..255
        myLds[(e >> 4) * KP_ + 80 + (e & 15)] = (_Float16)0.0f;
    }
    __syncthreads();

    // reload spectrum as A fragments: M=token, K=freq column c
    v16h sfrag[3];
#pragma unroll
    for (int s = 0; s < 3; ++s) {
#pragma unroll
        for (int g = 0; g < 2; ++g) {
            const int c0 = s * 32 + hi * 16 + g * 8;
#pragma unroll
            for (int w = 0; w < 8; ++w)
                sfrag[s][g * 8 + w] = myLds[tok * KP_ + c0 + w];
        }
    }

    // synthesis: Out^T(16x64) = Fg^T(16x96) * S(96x64)
    float* orow = out + ((size_t)(b * N_ + n0)) * D_ + h * HD_;
#pragma unroll
    for (int jt = 0; jt < 4; ++jt) {
        v8f c = {};
#pragma unroll
        for (int s = 0; s < 3; ++s) {
            v16h bfr = *(const v16h*)(syFrag + (size_t)(jt * 3 + s) * 512 + lane * 16);
            c = __builtin_amdgcn_wmma_f32_16x16x32_f16(false, sfrag[s], false, bfr,
                                                       (short)0, c, false, false);
        }
        const int j = jt * 16 + tok;
#pragma unroll
        for (int r = 0; r < 8; ++r) {
            const int mrow = r + hi * 8;
            orow[(size_t)mrow * D_ + j] = c[r];
        }
    }
}

extern "C" void kernel_launch(void* const* d_in, const int* in_sizes, int n_in,
                              void* d_out, int out_size, void* d_ws, size_t ws_size,
                              hipStream_t stream) {
    const float* x           = (const float*)d_in[0];
    const float* gamma       = (const float*)d_in[1];
    const float* beta        = (const float*)d_in[2];
    const float* base_filter = (const float*)d_in[3];
    const float* base_bias   = (const float*)d_in[4];
    const float* W1          = (const float*)d_in[5];
    const float* b1          = (const float*)d_in[6];
    const float* W2          = (const float*)d_in[7];
    const float* b2          = (const float*)d_in[8];
    float* out = (float*)d_out;

    char* ws = (char*)d_ws;
    float* mu      = (float*)(ws);                         // 32768 f32
    float* rstd    = (float*)(ws + 131072);                // 32768 f32
    float* ctxsum  = (float*)(ws + 262144);                // 8192 f32
    float* hmid    = (float*)(ws + 294912);                // 8192 f32
    float* filt    = (float*)(ws + 327680);                // 4224 f32
    float* biasEff = (float*)(ws + 344576);                // 4224 f32
    _Float16* gaFrag = (_Float16*)(ws + 361472);           // 5120 f16
    _Float16* syFrag = (_Float16*)(ws + 371712);           // 6144 f16

    zero_f32<<<32, 256, 0, stream>>>(ctxsum, B_ * D_);
    dft_setup_kernel<<<44, 256, 0, stream>>>(gaFrag, syFrag);
    ln_stats_kernel<<<B_ * (N_ / 64), 256, 0, stream>>>(x, mu, rstd, ctxsum);
    ctx_mlp1_kernel<<<(B_ * D_) / 256, 256, 0, stream>>>(ctxsum, gamma, beta, W1, b1, hmid);
    ctx_mlp2_kernel<<<(B_ * H_ * FB_ + 255) / 256, 256, 0, stream>>>(
        hmid, W2, b2, base_filter, base_bias, filt, biasEff);
    spectral_kernel<<<(B_ * H_ * (N_ / 16)) / 4, 128, 0, stream>>>(
        x, gamma, beta, mu, rstd, filt, biasEff, gaFrag, syFrag, out);
}
